// EquivariantGNN_50637664420143
// MI455X (gfx1250) — compile-verified, matrix operand order checked by reading, not run on hardware
//
#include <hip/hip_runtime.h>
#include <hip/hip_bf16.h>

typedef __attribute__((ext_vector_type(16))) _Float16 v16h;
typedef __attribute__((ext_vector_type(8)))  float    v8f;

#define NNODES 50000
#define NEDGES 800000
#define HDIM   128
#define NGRAPH 32

static __device__ __forceinline__ void atomic_add_f32(float* p, float v) {
    (void)__hip_atomic_fetch_add(p, v, __ATOMIC_RELAXED, __HIP_MEMORY_SCOPE_AGENT);
}

// ---------------- elementwise helpers ----------------

__global__ void zero_f32_kernel(float* __restrict__ p, int n) {
    int i = blockIdx.x * blockDim.x + threadIdx.x;
    if (i < n) p[i] = 0.0f;
}

__global__ void f32_to_f16_kernel(const float* __restrict__ X, _Float16* __restrict__ Xh, int n) {
    int i = blockIdx.x * blockDim.x + threadIdx.x;
    if (i < n) Xh[i] = (_Float16)X[i];
}

// W: [K,128] row-major fp32  ->  Wt: [128,K] row-major f16 (transposed, for B-fragment b128 loads)
__global__ void transpose_f32_to_f16_kernel(const float* __restrict__ W, _Float16* __restrict__ Wt, int K) {
    int i = blockIdx.x * blockDim.x + threadIdx.x;
    if (i >= K * HDIM) return;
    int k = i >> 7;     // i / 128
    int n = i & 127;    // i % 128
    Wt[(size_t)n * K + k] = (_Float16)W[i];
}

// U[n][0:128] = Xh[n], U[n][128:256] = (f16)aggr[n]
__global__ void build_U_kernel(const _Float16* __restrict__ Xh, const float* __restrict__ aggr,
                               _Float16* __restrict__ U, int n) {
    int i = blockIdx.x * blockDim.x + threadIdx.x;
    if (i >= n) return;
    int node = i >> 7, c = i & 127;
    U[(size_t)node * 256 + c]       = Xh[i];
    U[(size_t)node * 256 + 128 + c] = (_Float16)aggr[i];
}

// ---------------- WMMA GEMM: C[M,128] = act(A[M,K] @ Bt^T) (+bias, relu, +addsrc) ----------------
// A: f16 row-major [M,K]; Bt: f16 [128,K] (i.e. W transposed). One block = 16 rows,
// 8 waves = 8 column tiles of 16. Fragments per ISA 7.12.2 wave32 layouts.
__global__ __launch_bounds__(256) void gemm_wmma_kernel(
    const _Float16* __restrict__ A, const _Float16* __restrict__ Bt,
    const float* __restrict__ bias, const float* __restrict__ addsrc,
    float* __restrict__ C, int M, int K, int doRelu)
{
    (void)M;
    const int wave  = threadIdx.x >> 5;   // 0..7 -> column tile
    const int lane  = threadIdx.x & 31;
    const int n16   = lane & 15;          // row (A) / col (B) within tile
    const int half_ = lane >> 4;          // selects K-half of fragment
    const _Float16* Arow = A  + (size_t)(blockIdx.x * 16 + n16) * K;
    const _Float16* Brow = Bt + (size_t)(wave * 16 + n16) * K;

    union Frag { v16h h; uint4 q[2]; };
    v8f acc = {};
    for (int k0 = 0; k0 < K; k0 += 32) {
        Frag fa, fb;
        fa.q[0] = *(const uint4*)(Arow + k0 +      half_ * 8);
        fa.q[1] = *(const uint4*)(Arow + k0 + 16 + half_ * 8);
        fb.q[0] = *(const uint4*)(Brow + k0 +      half_ * 8);
        fb.q[1] = *(const uint4*)(Brow + k0 + 16 + half_ * 8);
        acc = __builtin_amdgcn_wmma_f32_16x16x32_f16(
            /*neg_a=*/false, fa.h, /*neg_b=*/false, fb.h,
            /*c_mod=*/(short)0, acc, /*reuse_a=*/false, /*reuse_b=*/false);
    }

    const int col = wave * 16 + n16;
    #pragma unroll
    for (int r = 0; r < 8; ++r) {
        const size_t row = (size_t)blockIdx.x * 16 + r + 8 * half_;
        float v = acc[r];
        if (bias)   v += bias[col];
        if (doRelu) v = fmaxf(v, 0.0f);
        if (addsrc) v += addsrc[row * HDIM + col];
        C[row * HDIM + col] = v;
    }
}

// ---------------- edge phase: msg = relu(A[row]+B[col]+dist*w3+b); aggr[col] += msg ----------------
// one wave per edge, each lane handles 4 channels (float4).
__global__ __launch_bounds__(256) void edge_msg_kernel(
    const float* __restrict__ A, const float* __restrict__ Bm,
    const float* __restrict__ w3, const float* __restrict__ bmsg,
    const float* __restrict__ edge_dist,
    const int* __restrict__ rowi, const int* __restrict__ coli,
    float* __restrict__ aggr, int E)
{
    int e = blockIdx.x * 8 + (threadIdx.x >> 5);
    if (e >= E) return;
    int lane = threadIdx.x & 31;
    int r = rowi[e], c = coli[e];
    float d = edge_dist[e];

    float4 av = ((const float4*)(A  + (size_t)r * HDIM))[lane];
    float4 bv = ((const float4*)(Bm + (size_t)c * HDIM))[lane];
    float4 wv = ((const float4*)w3)[lane];
    float4 kv = ((const float4*)bmsg)[lane];

    float m0 = fmaxf(av.x + bv.x + d * wv.x + kv.x, 0.0f);
    float m1 = fmaxf(av.y + bv.y + d * wv.y + kv.y, 0.0f);
    float m2 = fmaxf(av.z + bv.z + d * wv.z + kv.z, 0.0f);
    float m3 = fmaxf(av.w + bv.w + d * wv.w + kv.w, 0.0f);

    float* dst = aggr + (size_t)c * HDIM + lane * 4;
    atomic_add_f32(dst + 0, m0);
    atomic_add_f32(dst + 1, m1);
    atomic_add_f32(dst + 2, m2);
    atomic_add_f32(dst + 3, m3);
}

// ---------------- per-graph sum pooling ----------------
__global__ void pool_kernel(const float* __restrict__ h, const int* __restrict__ batch,
                            float* __restrict__ g, int n) {
    int i = blockIdx.x * blockDim.x + threadIdx.x;
    if (i >= n) return;
    int node = i >> 7, c = i & 127;
    atomic_add_f32(&g[(size_t)batch[node] * HDIM + c], h[i]);
}

// ---------------- head: out[32] = relu(g@w1+b1)@w2 + b2 (tiny; single block) ----------------
__global__ __launch_bounds__(256) void head_kernel(
    const float* __restrict__ g, const float* __restrict__ w1, const float* __restrict__ b1,
    const float* __restrict__ w2, const float* __restrict__ b2, float* __restrict__ out)
{
    __shared__ float hid[NGRAPH * HDIM];
    for (int idx = threadIdx.x; idx < NGRAPH * HDIM; idx += 256) {
        int r = idx >> 7, c = idx & 127;
        float s = b1[c];
        for (int k = 0; k < HDIM; ++k) s += g[r * HDIM + k] * w1[k * HDIM + c];
        hid[idx] = fmaxf(s, 0.0f);
    }
    __syncthreads();
    if (threadIdx.x < NGRAPH) {
        int r = threadIdx.x;
        float s = b2[0];
        for (int c = 0; c < HDIM; ++c) s += hid[r * HDIM + c] * w2[c];
        out[r] = s;
    }
}

// ---------------- launch ----------------
extern "C" void kernel_launch(void* const* d_in, const int* in_sizes, int n_in,
                              void* d_out, int out_size, void* d_ws, size_t ws_size,
                              hipStream_t stream) {
    (void)in_sizes; (void)n_in; (void)out_size; (void)ws_size;

    const float* x         = (const float*)d_in[0];
    const float* edge_dist = (const float*)d_in[1];
    const int*   edge_idx  = (const int*)d_in[2];
    const int*   batch     = (const int*)d_in[3];
    const float* w_res[2]  = {(const float*)d_in[4],  (const float*)d_in[9]};
    const float* w_msg[2]  = {(const float*)d_in[5],  (const float*)d_in[10]};
    const float* b_msg[2]  = {(const float*)d_in[6],  (const float*)d_in[11]};
    const float* w_upd[2]  = {(const float*)d_in[7],  (const float*)d_in[12]};
    const float* b_upd[2]  = {(const float*)d_in[8],  (const float*)d_in[13]};
    const float* w_p1 = (const float*)d_in[14];
    const float* b_p1 = (const float*)d_in[15];
    const float* w_p2 = (const float*)d_in[16];
    const float* b_p2 = (const float*)d_in[17];

    char* ws = (char*)d_ws;
    size_t off = 0;
    auto carve = [&](size_t bytes) -> void* {
        void* p = ws + off;
        off += (bytes + 255) & ~(size_t)255;
        return p;
    };
    _Float16* Xh   = (_Float16*)carve((size_t)NNODES * HDIM * 2);
    float*    Abuf = (float*)   carve((size_t)NNODES * HDIM * 4);
    float*    Bbuf = (float*)   carve((size_t)NNODES * HDIM * 4);
    float*    aggr = (float*)   carve((size_t)NNODES * HDIM * 4);
    _Float16* U    = (_Float16*)carve((size_t)NNODES * 2 * HDIM * 2);
    float*    Rbuf = (float*)   carve((size_t)NNODES * HDIM * 4);
    float*    Y1   = (float*)   carve((size_t)NNODES * HDIM * 4);
    _Float16* W1t  = (_Float16*)carve((size_t)HDIM * HDIM * 2);
    _Float16* W2t  = (_Float16*)carve((size_t)HDIM * HDIM * 2);
    _Float16* Wrt  = (_Float16*)carve((size_t)HDIM * HDIM * 2);
    _Float16* Wut  = (_Float16*)carve((size_t)HDIM * 2 * HDIM * 2);
    float*    g    = (float*)   carve((size_t)NGRAPH * HDIM * 4);
    float*    Y2   = Abuf;   // layer-2 A is dead by the time the final GEMM writes Y2

    const int* rowi = edge_idx;
    const int* coli = edge_idx + NEDGES;

    const int ELT = NNODES * HDIM;                 // 6.4M
    dim3 blk(256);
    int gridElt  = (ELT + 255) / 256;
    int gridW128 = (HDIM * HDIM + 255) / 256;
    int gridW256 = (2 * HDIM * HDIM + 255) / 256;
    int gridGemm = NNODES / 16;                    // 3125 (exact)
    int gridEdge = (NEDGES + 7) / 8;

    const float* Xin = x;
    float* Yout[2] = {Y1, Y2};
    for (int L = 0; L < 2; ++L) {
        transpose_f32_to_f16_kernel<<<gridW128, blk, 0, stream>>>(w_msg[L],               W1t, HDIM);
        transpose_f32_to_f16_kernel<<<gridW128, blk, 0, stream>>>(w_msg[L] + HDIM * HDIM, W2t, HDIM);
        transpose_f32_to_f16_kernel<<<gridW128, blk, 0, stream>>>(w_res[L],               Wrt, HDIM);
        transpose_f32_to_f16_kernel<<<gridW256, blk, 0, stream>>>(w_upd[L],               Wut, 2 * HDIM);

        f32_to_f16_kernel<<<gridElt, blk, 0, stream>>>(Xin, Xh, ELT);

        gemm_wmma_kernel<<<gridGemm, blk, 0, stream>>>(Xh, W1t, nullptr, nullptr, Abuf, NNODES, HDIM, 0);
        gemm_wmma_kernel<<<gridGemm, blk, 0, stream>>>(Xh, W2t, nullptr, nullptr, Bbuf, NNODES, HDIM, 0);

        zero_f32_kernel<<<gridElt, blk, 0, stream>>>(aggr, ELT);
        edge_msg_kernel<<<gridEdge, blk, 0, stream>>>(Abuf, Bbuf,
                                                      w_msg[L] + 2 * HDIM * HDIM, b_msg[L],
                                                      edge_dist, rowi, coli, aggr, NEDGES);

        build_U_kernel<<<gridElt, blk, 0, stream>>>(Xh, aggr, U, ELT);
        gemm_wmma_kernel<<<gridGemm, blk, 0, stream>>>(Xh, Wrt, nullptr, nullptr, Rbuf, NNODES, HDIM, 0);
        // Y = relu(U @ w_upd + b_upd) + (X @ w_res)
        gemm_wmma_kernel<<<gridGemm, blk, 0, stream>>>(U, Wut, b_upd[L], Rbuf, Yout[L], NNODES, 2 * HDIM, 1);
        Xin = Yout[L];
    }

    zero_f32_kernel<<<(NGRAPH * HDIM + 255) / 256, blk, 0, stream>>>(g, NGRAPH * HDIM);
    pool_kernel<<<gridElt, blk, 0, stream>>>(Xin, batch, g, ELT);
    head_kernel<<<1, 256, 0, stream>>>(g, w_p1, b_p1, w_p2, b_p2, (float*)d_out);
}
